// JambaMoE_40561671144130
// MI455X (gfx1250) — compile-verified, matrix operand order checked by reading, not run on hardware
//
#include <hip/hip_runtime.h>
#include <hip/hip_bf16.h>

// ---------------------------------------------------------------------------
// JambaMoE (top-2, E=8) for MI455X / gfx1250, wave32 + WMMA bf16, gathered.
//
// T=4096 tokens, H=2048, I=5632, E=8. Weights stream once (~1.1 GB fp32 ->
// ~50us @ 23.3 TB/s). Top-2 routing => ~1024 tokens/expert; token lists are
// compacted so the matrix cores only do the 567 GFLOP that matter
// (v_wmma_f32_16x16x32_bf16, f32 accum). bf16 A-tiles are staged via the
// gfx1250 async-to-LDS path (ASYNCcnt); fp32 weights convert to bf16 in
// VGPRs while staging (v_cvt_pk_bf16_f32) with global_prefetch_b8 lookahead.
// ---------------------------------------------------------------------------

typedef __attribute__((ext_vector_type(16))) __bf16 v16bf;
typedef __attribute__((ext_vector_type(8)))  float  v8f;
typedef int v4i_gcc __attribute__((vector_size(16)));   // matches builtin param

static constexpr int T_TOK = 4096;   // B*S = 2*2048
static constexpr int H_DIM = 2048;
static constexpr int I_DIM = 5632;
static constexpr int N_EXP = 8;

// ------------------------------------------------ gfx1250 async-to-LDS path

__device__ __forceinline__ void async_b128(const void* g, void* l) {
#if defined(__has_builtin) && __has_builtin(__builtin_amdgcn_global_load_async_to_lds_b128)
  __builtin_amdgcn_global_load_async_to_lds_b128(
      (__attribute__((address_space(1))) v4i_gcc*)(uintptr_t)g,
      (__attribute__((address_space(3))) v4i_gcc*)(uint32_t)(uintptr_t)l,
      0, 0);
#else
  *(uint4*)l = *(const uint4*)g;     // synchronous fallback
#endif
}

__device__ __forceinline__ void wait_async() {
#if defined(__has_builtin) && __has_builtin(__builtin_amdgcn_s_wait_asynccnt)
  __builtin_amdgcn_s_wait_asynccnt(0);
#else
  asm volatile("s_wait_asynccnt 0" ::: "memory");
#endif
}

// ---------------------------------------------------------------- utilities

__global__ void zero_f32(float* __restrict__ p, int n) {
  int i = blockIdx.x * blockDim.x + threadIdx.x;
  if (i < n) p[i] = 0.f;
}

__global__ void cast_f32_bf16(const float* __restrict__ in,
                              __bf16* __restrict__ out, int n) {
  int i = (blockIdx.x * blockDim.x + threadIdx.x) * 4;
  if (i + 3 < n) {
    float4 f = *(const float4*)(in + i);
    out[i + 0] = (__bf16)f.x;
    out[i + 1] = (__bf16)f.y;
    out[i + 2] = (__bf16)f.z;
    out[i + 3] = (__bf16)f.w;
  }
}

// ---------------------------------------------------------------- router
// One block/token; wave w computes expert w's logit (wave32 shuffle
// reduction); thread 0 does softmax + top-2 and appends (token, prob) to the
// two winning experts' compacted lists. List order is atomic-nondeterministic
// but every per-token result downstream is position-independent, so the
// final output values are deterministic.

__global__ __launch_bounds__(256)
void moe_router(const float* __restrict__ x,
                const float* __restrict__ rw,
                int*   __restrict__ cnt,   // [E]
                int*   __restrict__ tok,   // [E*T]
                float* __restrict__ wt) {  // [E*T]
  const int t    = blockIdx.x;
  const int lane = threadIdx.x & 31;
  const int e    = threadIdx.x >> 5;          // 0..7

  const float* xr = x  + (size_t)t * H_DIM;
  const float* wr = rw + (size_t)e * H_DIM;

  float s = 0.f;
  for (int h = lane; h < H_DIM; h += 32) s += xr[h] * wr[h];
  #pragma unroll
  for (int off = 16; off > 0; off >>= 1) s += __shfl_xor(s, off, 32);

  __shared__ float logit[N_EXP];
  if (lane == 0) logit[e] = s;
  __syncthreads();

  if (threadIdx.x == 0) {
    float m = logit[0];
    #pragma unroll
    for (int i = 1; i < N_EXP; ++i) m = fmaxf(m, logit[i]);
    float p[N_EXP]; float sum = 0.f;
    #pragma unroll
    for (int i = 0; i < N_EXP; ++i) { p[i] = __expf(logit[i] - m); sum += p[i]; }
    float inv = 1.f / sum;
    #pragma unroll
    for (int i = 0; i < N_EXP; ++i) p[i] *= inv;
    int i1 = 0;
    #pragma unroll
    for (int i = 1; i < N_EXP; ++i) if (p[i] > p[i1]) i1 = i;
    int i2 = (i1 == 0) ? 1 : 0;
    #pragma unroll
    for (int i = 0; i < N_EXP; ++i) if (i != i1 && p[i] > p[i2]) i2 = i;

    int pos1 = atomicAdd(&cnt[i1], 1);
    tok[(size_t)i1 * T_TOK + pos1] = t;
    wt [(size_t)i1 * T_TOK + pos1] = p[i1];
    int pos2 = atomicAdd(&cnt[i2], 1);
    tok[(size_t)i2 * T_TOK + pos2] = t;
    wt [(size_t)i2 * T_TOK + pos2] = p[i2];
  }
}

// -------------------------------------------------- WMMA fragment helpers
// A/B operand (16x32 bf16, wave32): lane holds row (lane&15); lane>>4 picks
// the K half. 16 contiguous bf16 = 32B from LDS.

__device__ __forceinline__ v16bf frag_ld(const __bf16* lds, int row0, int lane) {
  return *(const v16bf*)(lds + (size_t)(row0 + (lane & 15)) * 32 + (lane >> 4) * 16);
}

__device__ __forceinline__ v8f wmma_bf16(v16bf a, v16bf b, v8f c) {
  return __builtin_amdgcn_wmma_f32_16x16x32_bf16(
      false, a, false, b, (short)0, c, false, false);
}

// ---------------------------------------------------------------- gate+up
// act[pos,i] = silu(x[tok[pos]] @ wg^T) * (x[tok[pos]] @ wu^T), bf16.
// Block tile 128(M=list positions) x 64(N), K-step 32 over H. 8 waves (4x2),
// each owns 32x32 with two accumulator sets. Worst-case grid; early-exit on
// the device-side expert count.

__global__ __launch_bounds__(256)
void gemm_gateup(const __bf16* __restrict__ xb,
                 const float*  __restrict__ wg,
                 const float*  __restrict__ wu,
                 __bf16* __restrict__ act,
                 const int* __restrict__ tok,
                 const int* __restrict__ cnt) {
  const int n  = *cnt;
  const int bm = blockIdx.y * 128;
  if (bm >= n) return;

  __shared__ __bf16 lA[128 * 32];
  __shared__ __bf16 lG[64 * 32];
  __shared__ __bf16 lU[64 * 32];

  const int tid  = threadIdx.x;
  const int lane = tid & 31;
  const int wave = tid >> 5;
  const int wm   = wave >> 1;       // 0..3 -> 32-row strip
  const int wn   = wave & 1;        // 0..1 -> 32-col strip
  const int bn   = blockIdx.x * 64;

  // gathered A row for this thread's staging slot (k-independent)
  const int ar = tid >> 1, ac = (tid & 1) * 16;
  const int apos = bm + ar;
  const int atok = tok[(apos < n) ? apos : (n - 1)];
  const __bf16* aSrcRow = xb + (size_t)atok * H_DIM + ac;

  v8f accG[2][2], accU[2][2];
  const v8f vzero = {0.f, 0.f, 0.f, 0.f, 0.f, 0.f, 0.f, 0.f};
  #pragma unroll
  for (int i = 0; i < 2; ++i)
    #pragma unroll
    for (int j = 0; j < 2; ++j) { accG[i][j] = vzero; accU[i][j] = vzero; }

  for (int k0 = 0; k0 < H_DIM; k0 += 32) {
    // --- stage A (bf16 gather): async global->LDS, 2x b128 per thread
    {
      __bf16* dst = &lA[ar * 32 + ac];
      async_b128(aSrcRow + k0, dst);
      async_b128(aSrcRow + k0 + 8, dst + 8);
    }
    // --- stage B gate/up (fp32 -> bf16): 64x32 each, half the block per tile
    {
      int q = tid & 127;
      int r = q >> 1, c = (q & 1) * 16;
      const float* src = ((tid < 128) ? wg : wu) + (size_t)(bn + r) * H_DIM + k0 + c;
      __bf16* dst = ((tid < 128) ? lG : lU) + r * 32 + c;
      #pragma unroll
      for (int v = 0; v < 4; ++v) {
        float4 f = ((const float4*)src)[v];
        dst[v * 4 + 0] = (__bf16)f.x; dst[v * 4 + 1] = (__bf16)f.y;
        dst[v * 4 + 2] = (__bf16)f.z; dst[v * 4 + 3] = (__bf16)f.w;
      }
      if (k0 + 32 < H_DIM)                       // global_prefetch_b8
        __builtin_prefetch(src + 32, 0, 1);
    }
    wait_async();
    __syncthreads();

    v16bf a[2], bg[2], bu[2];
    #pragma unroll
    for (int i = 0; i < 2; ++i) a[i]  = frag_ld(lA, wm * 32 + i * 16, lane);
    #pragma unroll
    for (int j = 0; j < 2; ++j) {
      bg[j] = frag_ld(lG, wn * 32 + j * 16, lane);
      bu[j] = frag_ld(lU, wn * 32 + j * 16, lane);
    }
    #pragma unroll
    for (int i = 0; i < 2; ++i)
      #pragma unroll
      for (int j = 0; j < 2; ++j) {
        accG[i][j] = wmma_bf16(a[i], bg[j], accG[i][j]);
        accU[i][j] = wmma_bf16(a[i], bu[j], accU[i][j]);
      }
    __syncthreads();
  }

  // Epilogue: SiLU(g)*u -> bf16 act (by list position).
  #pragma unroll
  for (int i = 0; i < 2; ++i)
    #pragma unroll
    for (int j = 0; j < 2; ++j)
      #pragma unroll
      for (int r = 0; r < 8; ++r) {
        int pos = bm + wm * 32 + i * 16 + (lane >> 4) * 8 + r;
        int col = bn + wn * 32 + j * 16 + (lane & 15);
        float g = accG[i][j][r], u = accU[i][j][r];
        float h = g * (1.f / (1.f + __expf(-g))) * u;
        act[(size_t)pos * I_DIM + col] = (__bf16)h;
      }
}

// ---------------------------------------------------------------- down-proj
// out[tok[pos],h] += wt[pos] * (act[pos] @ wd^T). Block 128(M) x 128(N over
// H), K-step 32 over I. 8 waves (4x2), each 32x64. out pre-zeroed; each pos
// is a unique token within a launch and expert launches are sequential.

__global__ __launch_bounds__(256)
void gemm_down(const __bf16* __restrict__ act,
               const float*  __restrict__ wd,
               const int*    __restrict__ tok,
               const float*  __restrict__ wt,
               const int*    __restrict__ cnt,
               float* __restrict__ out) {
  const int n  = *cnt;
  const int bm = blockIdx.y * 128;
  if (bm >= n) return;

  __shared__ __bf16 lA[128 * 32];
  __shared__ __bf16 lB[128 * 32];

  const int tid  = threadIdx.x;
  const int lane = tid & 31;
  const int wave = tid >> 5;
  const int wm   = wave >> 1;       // 0..3 -> 32 rows
  const int wn   = wave & 1;        // 0..1 -> 64 cols
  const int bn   = blockIdx.x * 128;

  const int ar = tid >> 1, ac = (tid & 1) * 16;
  const __bf16* aSrcRow = act + (size_t)(bm + ar) * I_DIM + ac;

  v8f acc[2][4];
  const v8f vzero = {0.f, 0.f, 0.f, 0.f, 0.f, 0.f, 0.f, 0.f};
  #pragma unroll
  for (int i = 0; i < 2; ++i)
    #pragma unroll
    for (int j = 0; j < 4; ++j) acc[i][j] = vzero;

  for (int k0 = 0; k0 < I_DIM; k0 += 32) {
    {   // stage A from bf16 act: async global->LDS
      __bf16* dst = &lA[ar * 32 + ac];
      async_b128(aSrcRow + k0, dst);
      async_b128(aSrcRow + k0 + 8, dst + 8);
    }
    {   // stage B from fp32 wd (row h, K = I), convert to bf16
      int r = tid >> 1, c = (tid & 1) * 16;
      const float* src = wd + (size_t)(bn + r) * I_DIM + k0 + c;
      __bf16* dst = lB + r * 32 + c;
      #pragma unroll
      for (int v = 0; v < 4; ++v) {
        float4 f = ((const float4*)src)[v];
        dst[v * 4 + 0] = (__bf16)f.x; dst[v * 4 + 1] = (__bf16)f.y;
        dst[v * 4 + 2] = (__bf16)f.z; dst[v * 4 + 3] = (__bf16)f.w;
      }
      if (k0 + 32 < I_DIM)
        __builtin_prefetch(src + 32, 0, 1);
    }
    wait_async();
    __syncthreads();

    v16bf a[2], b[4];
    #pragma unroll
    for (int i = 0; i < 2; ++i) a[i] = frag_ld(lA, wm * 32 + i * 16, lane);
    #pragma unroll
    for (int j = 0; j < 4; ++j) b[j] = frag_ld(lB, wn * 64 + j * 16, lane);
    #pragma unroll
    for (int i = 0; i < 2; ++i)
      #pragma unroll
      for (int j = 0; j < 4; ++j)
        acc[i][j] = wmma_bf16(a[i], b[j], acc[i][j]);
    __syncthreads();
  }

  #pragma unroll
  for (int i = 0; i < 2; ++i)
    #pragma unroll
    for (int j = 0; j < 4; ++j)
      #pragma unroll
      for (int r = 0; r < 8; ++r) {
        int pos = bm + wm * 32 + i * 16 + (lane >> 4) * 8 + r;
        if (pos < n) {
          int   t = tok[pos];
          float w = wt[pos];
          int col = bn + wn * 64 + j * 16 + (lane & 15);
          size_t idx = (size_t)t * H_DIM + col;
          out[idx] += w * acc[i][j][r];
        }
      }
}

// ---------------------------------------------------------------- launch

extern "C" void kernel_launch(void* const* d_in, const int* in_sizes, int n_in,
                              void* d_out, int out_size, void* d_ws, size_t ws_size,
                              hipStream_t stream) {
  const float* x  = (const float*)d_in[0];   // [T, H]
  const float* rw = (const float*)d_in[1];   // [E, H]
  const float* wg = (const float*)d_in[2];   // [E, I, H]
  const float* wu = (const float*)d_in[3];   // [E, I, H]
  const float* wd = (const float*)d_in[4];   // [E, H, I]
  float* out = (float*)d_out;                // [T, H]

  // workspace layout
  const size_t XB_BYTES  = (size_t)T_TOK * H_DIM * sizeof(__bf16);  // 16 MB
  const size_t CNT_BYTES = 256;                                     // E ints, padded
  const size_t TOK_BYTES = (size_t)N_EXP * T_TOK * sizeof(int);     // 128 KB
  const size_t WT_BYTES  = (size_t)N_EXP * T_TOK * sizeof(float);   // 128 KB
  char* ws = (char*)d_ws;
  __bf16* xb   = (__bf16*)ws;
  int*    cnt  = (int*)  (ws + XB_BYTES);
  int*    tok  = (int*)  (ws + XB_BYTES + CNT_BYTES);
  float*  wt   = (float*)(ws + XB_BYTES + CNT_BYTES + TOK_BYTES);
  __bf16* actb = (__bf16*)(ws + XB_BYTES + CNT_BYTES + TOK_BYTES + WT_BYTES);

  // 0) zero accumulator output and expert counters (buffers arrive poisoned)
  zero_f32<<<(T_TOK * H_DIM + 255) / 256, 256, 0, stream>>>(out, T_TOK * H_DIM);
  zero_f32<<<1, 256, 0, stream>>>((float*)cnt, N_EXP);

  // 1) cast tokens to bf16
  {
    int nel = T_TOK * H_DIM;
    cast_f32_bf16<<<nel / (256 * 4), 256, 0, stream>>>(x, xb, nel);
  }
  // 2) router -> compacted per-expert (token, weight) lists
  moe_router<<<T_TOK, 256, 0, stream>>>(x, rw, cnt, tok, wt);

  // 3) per-expert SwiGLU MLP on matrix cores, gathered tokens only.
  //    Worst-case grids; blocks early-exit on the device-side count.
  for (int e = 0; e < N_EXP; ++e) {
    const float* wge = wg + (size_t)e * I_DIM * H_DIM;
    const float* wue = wu + (size_t)e * I_DIM * H_DIM;
    const float* wde = wd + (size_t)e * H_DIM * I_DIM;
    const int*   te  = tok + (size_t)e * T_TOK;
    const float* we  = wt  + (size_t)e * T_TOK;
    gemm_gateup<<<dim3(I_DIM / 64, T_TOK / 128), 256, 0, stream>>>(
        xb, wge, wue, actb, te, cnt + e);
    gemm_down<<<dim3(H_DIM / 128, T_TOK / 128), 256, 0, stream>>>(
        actb, wde, te, we, cnt + e, out);
  }
}
